// VCDN_9723805958440
// MI455X (gfx1250) — matrix operand order repaired and checked by reading, not compiled, not run on hardware
//
#include <hip/hip_runtime.h>
#include <hip/hip_bf16.h>

#define FDIM   100
#define JK     10000
#define KCH    64                 // K per main-loop iteration (2 WMMA K-steps)
#define YS     10048              // padded y row stride (bf16 elems) = 157*64, zero tail
#define HDIM   64
#define BATCH  128
#define NSLAB  4
#define NCHUNK 157                // ceil(10000/64)
#define NBLK   (FDIM * NSLAB)     // 400 workgroups in main GEMM

typedef __attribute__((ext_vector_type(16))) __bf16   v16bf;
typedef __attribute__((ext_vector_type(8)))  float    v8f;
typedef __attribute__((ext_vector_type(4)))  float    v4f;
typedef __attribute__((ext_vector_type(4)))  unsigned v4u;

union Frag { v4u u[2]; v16bf v; };   // 32 bytes = 8 VGPRs = one 16x32 bf16 fragment

__device__ __forceinline__ unsigned f2bf_rne(float f) {
  unsigned u = __float_as_uint(f);
  return (u + 0x7fffu + ((u >> 16) & 1u)) >> 16;   // round-to-nearest-even
}

#if __has_builtin(__builtin_amdgcn_cvt_pk_bf16_f32)
__device__ __forceinline__ unsigned pk_bf16(float lo, float hi) {
  auto r = __builtin_amdgcn_cvt_pk_bf16_f32(lo, hi);   // v_cvt_pk_bf16_f32
  return __builtin_bit_cast(unsigned, r);
}
#else
__device__ __forceinline__ unsigned pk_bf16(float lo, float hi) {
  return (f2bf_rne(lo) & 0xffffu) | (f2bf_rne(hi) << 16);
}
#endif

// ---------------- Kernel 1: y[b, jk] = bf16(x1[b,j] * x2[b,k]), zero-padded rows -------------
__global__ __launch_bounds__(256) void vcdn_y_kernel(const float* __restrict__ x,
                                                     unsigned short* __restrict__ y) {
  int idx = blockIdx.x * 256 + threadIdx.x;
  if (idx >= BATCH * YS) return;
  int b  = idx / YS;
  int jk = idx - b * YS;
  unsigned short r = 0;
  if (jk < JK) {
    int j = jk / FDIM;
    int k = jk - j * FDIM;
    float v = x[(size_t)b * 300 + 100 + j] * x[(size_t)b * 300 + 200 + k];
    r = (unsigned short)f2bf_rne(v);
  }
  y[idx] = r;
}

// ------------------- Main GEMM helpers: software-pipelined K=64 stages ----------------------
struct Pref {                       // one iteration's prefetched operands (regs)
  v4f  w0[2], w1[2];                // W1 fp32: [sub k32] two rows of a k-pair, 4 cols
  Frag a0[2], a1[2];                // A frags: [sub k32] for the wave's two M-tiles
};

__device__ __forceinline__ void prefetch_chunk(int jk0, const float* __restrict__ wbase,
                                               const unsigned short* __restrict__ yrow0,
                                               const unsigned short* __restrict__ yrow1,
                                               int kp, int n0, int lh, Pref& p) {
  // W1 first (pack depends on these), A after (only WMMA depends on them)
#pragma unroll
  for (int s = 0; s < 2; ++s) {
    int krow = jk0 + (s << 5) + (kp << 1);
    krow = krow < (JK - 2) ? krow : (JK - 2);        // clamp; padded-y A rows are 0 there
    const v4f* wp = (const v4f*)(wbase + (size_t)krow * HDIM + n0);
    p.w0[s] = __builtin_nontemporal_load(wp);
    p.w1[s] = __builtin_nontemporal_load(wp + (HDIM / 4));
  }
#pragma unroll
  for (int s = 0; s < 2; ++s) {
    const int seg = jk0 + (s << 5) + (lh << 4);
    p.a0[s].u[0] = *(const v4u*)(yrow0 + seg);
    p.a0[s].u[1] = *(const v4u*)(yrow0 + seg + 8);
    p.a1[s].u[0] = *(const v4u*)(yrow1 + seg);
    p.a1[s].u[1] = *(const v4u*)(yrow1 + seg + 8);
  }
}

// ---------------- Kernel 2: main GEMM, partial[b,h] = x0[b,i] * (y @ W1_i) ------------------
// Grid: 400 blocks (i*4 + slab). Block: 256 threads = 8 waves as 4(M) x 2(N).
__global__ __launch_bounds__(256) void vcdn_main_kernel(const float* __restrict__ x,
                                                        const float* __restrict__ W1,
                                                        const unsigned short* __restrict__ y,
                                                        float* __restrict__ partials) {
  __shared__ unsigned sB[2][2][1024];   // [buf][sub k32][frag-order dwords] = 16KB

  const int blk  = blockIdx.x;
  const int i    = blk >> 2;
  const int slab = blk & 3;
  const int c0 = (NCHUNK * slab) >> 2;
  const int c1 = (NCHUNK * (slab + 1)) >> 2;

  const int t  = threadIdx.x;
  const int w  = t >> 5, l = t & 31;
  const int mg = w & 3,  ng = w >> 2;
  const int lm = l & 15, lh = l >> 4;

  const float* wbase = W1 + (size_t)i * JK * HDIM;
  const unsigned short* yrow0 = y + (size_t)(32 * mg + lm) * YS;   // M-tile 2*mg
  const unsigned short* yrow1 = yrow0 + (size_t)16 * YS;           // M-tile 2*mg+1

  // staging map: thread t packs k-pair kp for 4 columns starting at n0 (per 32-K sub-chunk)
  const int kp = t >> 4;
  const int n0 = (t & 15) << 2;
  const int soff = ((n0 >> 4) << 8) + (((n0 & 15) + ((kp >> 3) << 4)) << 3) + (kp & 7);
  const int boff0 = ((2 * ng) << 8) + l * 8;        // B frag read offsets (dwords)
  const int boff1 = boff0 + 256;

  v8f acc00 = {}, acc01 = {}, acc10 = {}, acc11 = {};

  Pref cur, nxt;
  prefetch_chunk(c0 << 6, wbase, yrow0, yrow1, kp, n0, lh, cur);

  int buf = 0;
#pragma unroll 2
  for (int c = c0; c < c1; ++c) {
    // ---- stage current chunk's W1 (fp32 -> packed bf16) into LDS[buf] ----
    unsigned* sbuf = &sB[buf][0][0];
#pragma unroll
    for (int s = 0; s < 2; ++s) {
      unsigned* d = sbuf + (s << 10) + soff;
      d[0]  = pk_bf16(cur.w0[s].x, cur.w1[s].x);
      d[8]  = pk_bf16(cur.w0[s].y, cur.w1[s].y);
      d[16] = pk_bf16(cur.w0[s].z, cur.w1[s].z);
      d[24] = pk_bf16(cur.w0[s].w, cur.w1[s].w);
    }

    // ---- L2 prefetch of the W1 stream ~4 chunks (64KB) ahead ----
    {
      int jkp = (c << 6) + 4 * KCH;
      jkp = jkp < (JK - KCH) ? jkp : (JK - KCH);
      __builtin_prefetch(wbase + (size_t)jkp * HDIM + t * 16, 0, 0);
    }

    // ---- issue next chunk's global loads (in flight across barrier + WMMAs) ----
    const int cn = (c + 1 < c1) ? (c + 1) : c;
    prefetch_chunk(cn << 6, wbase, yrow0, yrow1, kp, n0, lh, nxt);

    __syncthreads();   // backend drains dscnt to 0 before signal -> single barrier is safe

    // ---- consume LDS[buf]: 2 sub-chunks x 4 WMMAs ----
#pragma unroll
    for (int s = 0; s < 2; ++s) {
      const unsigned* sbase = sbuf + (s << 10);
      Frag b0, b1;
      b0.u[0] = *(const v4u*)(sbase + boff0);
      b0.u[1] = *(const v4u*)(sbase + boff0 + 4);
      b1.u[0] = *(const v4u*)(sbase + boff1);
      b1.u[1] = *(const v4u*)(sbase + boff1 + 4);
      acc00 = __builtin_amdgcn_wmma_f32_16x16x32_bf16(false, cur.a0[s].v, false, b0.v, (short)0, acc00, false, false);
      acc01 = __builtin_amdgcn_wmma_f32_16x16x32_bf16(false, cur.a0[s].v, false, b1.v, (short)0, acc01, false, false);
      acc10 = __builtin_amdgcn_wmma_f32_16x16x32_bf16(false, cur.a1[s].v, false, b0.v, (short)0, acc10, false, false);
      acc11 = __builtin_amdgcn_wmma_f32_16x16x32_bf16(false, cur.a1[s].v, false, b1.v, (short)0, acc11, false, false);
    }

    buf ^= 1;
    cur = nxt;        // renamed away under unroll-by-2
  }

  // ---- epilogue: fold x0[b,i] in fp32, write this block's deterministic partial ----
  float* pblk = partials + (size_t)blk * (BATCH * HDIM);
  const int ncol = 32 * ng + lm;
#pragma unroll
  for (int g = 0; g < 8; ++g) {
    int br0 = 32 * mg + (lh << 3) + g;        // rows of M-tile 2*mg
    int br1 = br0 + 16;                       // rows of M-tile 2*mg+1
    float s0 = x[(size_t)br0 * 300 + i];      // x[b, 0, i]
    float s1 = x[(size_t)br1 * 300 + i];
    pblk[br0 * HDIM + ncol]      = s0 * acc00[g];
    pblk[br0 * HDIM + ncol + 16] = s0 * acc01[g];
    pblk[br1 * HDIM + ncol]      = s1 * acc10[g];
    pblk[br1 * HDIM + ncol + 16] = s1 * acc11[g];
  }
}

// ---------------- Kernel 3: reduce 400 partials, +b1, ReLU, tiny 64x5 head ------------------
__global__ __launch_bounds__(128) void vcdn_head_kernel(const float* __restrict__ partials,
                                                        const float* __restrict__ b1,
                                                        const float* __restrict__ W2,
                                                        const float* __restrict__ b2,
                                                        float* __restrict__ out) {
  __shared__ float hsh[128];
  const int t = threadIdx.x;
  const int base = blockIdx.x * 128;          // covers 2 batches x 64 hidden
  float s = 0.f;
#pragma unroll 4
  for (int p = 0; p < NBLK; ++p)
    s += partials[(size_t)p * (BATCH * HDIM) + base + t];
  s += b1[t & 63];
  s = fmaxf(s, 0.f);
  hsh[t] = s;
  __syncthreads();
  if (t < 10) {
    int bl = t / 5, c = t - bl * 5;
    const float* hv = &hsh[bl * 64];
    float o = b2[c];
#pragma unroll
    for (int q = 0; q < 64; ++q) o += hv[q] * W2[q * 5 + c];
    out[(blockIdx.x * 2 + bl) * 5 + c] = o;
  }
}

extern "C" void kernel_launch(void* const* d_in, const int* in_sizes, int n_in,
                              void* d_out, int out_size, void* d_ws, size_t ws_size,
                              hipStream_t stream) {
  const float* x  = (const float*)d_in[0];
  const float* W1 = (const float*)d_in[1];
  const float* b1 = (const float*)d_in[2];
  const float* W2 = (const float*)d_in[3];
  const float* b2 = (const float*)d_in[4];
  float* out = (float*)d_out;

  // workspace: [0, 13.1MB) per-block partials; then padded bf16 y (~2.57MB)
  float* partials = (float*)d_ws;
  unsigned short* y =
      (unsigned short*)((char*)d_ws + (size_t)NBLK * BATCH * HDIM * sizeof(float));

  vcdn_y_kernel<<<(BATCH * YS + 255) / 256, 256, 0, stream>>>(x, y);
  vcdn_main_kernel<<<NBLK, 256, 0, stream>>>(x, W1, y, partials);
  vcdn_head_kernel<<<(BATCH * HDIM) / 128, 128, 0, stream>>>(partials, b1, W2, b2, out);
}
